// disparity_48808008352330
// MI455X (gfx1250) — matrix operand order repaired and checked by reading, not compiled, not run on hardware
//
#include <hip/hip_runtime.h>

// ---------------------------------------------------------------------------
// Prototype-loss on MI455X (gfx1250, wave32, WMMA).
// Segment-sum as GEMM:  sums[m,n] = sum_k onehot(mask_k)[m] * feat_k[n]
//   A: 16x32 f16 one-hot of class ids (exact in f16)
//   B: 32x16 f16 feats (cols 0..8), col 9 = 1.0 (yields counts), cols 10..15 = 0
//   C/D: 16x16 f32 accumulator, one v_wmma_f32_16x16x32_f16 per 32 pixels.
// Memory bound: ~189 MB @ 23.3 TB/s => ~8 us floor; WMMA removes all scalar
// accumulation cost so the kernel stays on the bandwidth roof.
// ---------------------------------------------------------------------------

typedef __attribute__((ext_vector_type(16))) _Float16 v16h;
typedef __attribute__((ext_vector_type(8)))  float    v8f;

#define NCLS 9
#define NB   8
#define NH   768
#define NW   768
#define HW   (NH * NW)            // 589824 pixels per batch
#define CHUNKS (HW / 32)          // 18432 chunks of 32 pixels
#define WG_PER_BATCH 32
#define WAVES_PER_WG 8            // 256 threads = 8 wave32
#define WAVES_PB (WG_PER_BATCH * WAVES_PER_WG)   // 256 waves per batch

__global__ void zero_ws_kernel(float* __restrict__ ws, int nel) {
  int i = blockIdx.x * blockDim.x + threadIdx.x;
  if (i < nel) ws[i] = 0.0f;
}

__global__ __launch_bounds__(256)
void proto_accum_kernel(const int*   __restrict__ masks,
                        const float* __restrict__ outputs,
                        float*       __restrict__ ws) {
  __shared__ float lds[256];

  const int tid  = threadIdx.x;
  const int lane = tid & 31;
  const int wave = tid >> 5;
  const int b       = blockIdx.x / WG_PER_BATCH;
  const int wg_in_b = blockIdx.x % WG_PER_BATCH;
  const int waveId  = wg_in_b * WAVES_PER_WG + wave;

  lds[tid] = 0.0f;
  __syncthreads();

  const int row   = lane & 15;       // A-matrix row (class id this lane tests)
  const int hi    = lane >> 4;       // lane half
  const int koffA = hi ? 8  : 0;     // A-matrix K offset per ISA layout
  const int koffB = hi ? 8  : 0;     // B-matrix K offset per ISA layout
  const int n     = lane & 15;       // B column = channel (9=count col)

  const int*   mbase  = masks + (size_t)b * HW;
  const float* fplane = outputs + ((size_t)b * NCLS + n) * (size_t)HW; // only deref if n<9

  // Constant B for non-channel columns: col 9 = 1.0 (count), 10..15 = 0.
  v16h bv;
  {
    _Float16 fill = (n == 9) ? (_Float16)1.0f : (_Float16)0.0f;
#pragma unroll
    for (int e = 0; e < 16; ++e) bv[e] = fill;
  }

  v8f acc = {};

  for (int chunk = waveId; chunk < CHUNKS; chunk += WAVES_PB) {
    const int base = chunk * 32;

    // ---- A matrix: one-hot of 16 mask values this lane owns -------------
    const int* mp = mbase + base + koffA;
    int4 ma0 = *(const int4*)(mp);
    int4 ma1 = *(const int4*)(mp + 4);
    int4 mb0 = *(const int4*)(mp + 16);
    int4 mb1 = *(const int4*)(mp + 20);
    int mk[16] = { ma0.x, ma0.y, ma0.z, ma0.w,  ma1.x, ma1.y, ma1.z, ma1.w,
                   mb0.x, mb0.y, mb0.z, mb0.w,  mb1.x, mb1.y, mb1.z, mb1.w };
    v16h a;
#pragma unroll
    for (int e = 0; e < 16; ++e)
      a[e] = (mk[e] == row) ? (_Float16)1.0f : (_Float16)0.0f;

    // ---- B matrix: feats k=8h..8h+7 and 16+8h..16+8h+7 of this lane's channel plane --
    if (n < NCLS) {
      const float4* fp = (const float4*)(fplane + base + koffB);
      float4 f0 = fp[0], f1 = fp[1], f2 = fp[4], f3 = fp[5];
      float ff[16] = { f0.x, f0.y, f0.z, f0.w,  f1.x, f1.y, f1.z, f1.w,
                       f2.x, f2.y, f2.z, f2.w,  f3.x, f3.y, f3.z, f3.w };
#pragma unroll
      for (int e = 0; e < 16; ++e) bv[e] = (_Float16)ff[e];
      // Speculative streaming prefetch one wave-stride ahead (dropped on OOB).
      __builtin_prefetch(fplane + base + koffB + WAVES_PB * 32, 0, 1);
    }

    // ---- 32 pixels reduced per instruction ------------------------------
    acc = __builtin_amdgcn_wmma_f32_16x16x32_f16(
        /*neg_a=*/false, a, /*neg_b=*/false, bv,
        /*c_mod=*/(short)0, acc, /*reuse_a=*/false, /*reuse_b=*/false);
  }

  // ---- combine the workgroup's 8 waves in LDS (ds_add_f32) --------------
  const int M0 = hi * 8;   // D layout: VGPR r -> M = r + (lane<16 ? 0 : 8)
#pragma unroll
  for (int r = 0; r < 8; ++r)
    atomicAdd(&lds[(M0 + r) * 16 + n], acc[r]);
  __syncthreads();

  // ---- one global atomic per thread into ws[b][16][16] ------------------
  atomicAdd(&ws[b * 256 + tid], lds[tid]);
}

__global__ __launch_bounds__(64)
void finalize_kernel(const float* __restrict__ ws, float* __restrict__ out) {
  __shared__ float s_loss[64];
  __shared__ float s_pres[64];
  const int tid = threadIdx.x;
  const int b = tid >> 3;
  const int k = (tid & 7) + 1;            // classes 1..8

  const float* row = ws + b * 256 + k * 16;
  float count = row[9];
  float denom = fmaxf(count, 1.0f);

  float p[NCLS];
  float mx = -INFINITY;
#pragma unroll
  for (int c = 0; c < NCLS; ++c) { p[c] = row[c] / denom; mx = fmaxf(mx, p[c]); }
  float sum = 0.0f;
#pragma unroll
  for (int c = 0; c < NCLS; ++c) sum += __expf(p[c] - mx);
  float lse = __logf(sum) + mx;

  float S = 0.0f;
#pragma unroll
  for (int c = 0; c < NCLS; ++c) S += (p[c] - lse);
  float lk = p[k] - lse;
  // tgt row: 0.9 at c==k, 0.0125 elsewhere  =>  per = -(0.8875*logp[k] + 0.0125*S)
  float per = -(0.8875f * lk + 0.0125f * S);

  bool present = count > 0.0f;
  s_loss[tid] = present ? per : 0.0f;
  s_pres[tid] = present ? 1.0f : 0.0f;
  __syncthreads();
  for (int off = 32; off > 0; off >>= 1) {
    if (tid < off) { s_loss[tid] += s_loss[tid + off]; s_pres[tid] += s_pres[tid + off]; }
    __syncthreads();
  }
  if (tid == 0) out[0] = s_loss[0] / fmaxf(s_pres[0], 1.0f);
}

extern "C" void kernel_launch(void* const* d_in, const int* in_sizes, int n_in,
                              void* d_out, int out_size, void* d_ws, size_t ws_size,
                              hipStream_t stream) {
  const int*   masks   = (const int*)d_in[0];     // (8, 768, 768) int32
  const float* outputs = (const float*)d_in[1];   // (8, 9, 768, 768) f32
  float* ws  = (float*)d_ws;                      // 8 * 16 * 16 f32 = 8 KB
  float* out = (float*)d_out;                     // scalar f32

  zero_ws_kernel<<<8, 256, 0, stream>>>(ws, NB * 256);
  proto_accum_kernel<<<NB * WG_PER_BATCH, 256, 0, stream>>>(masks, outputs, ws);
  finalize_kernel<<<1, 64, 0, stream>>>(ws, out);
}
